// TransformerBlock_86131274154184
// MI455X (gfx1250) — matrix operand order repaired and checked by reading, not compiled, not run on hardware
//
#include <hip/hip_runtime.h>
#include <hip/hip_bf16.h>

// ---------------------------------------------------------------------------
// Types for CDNA5 WMMA (wave32, 16x16x32 bf16 -> f32)
// ---------------------------------------------------------------------------
typedef __attribute__((ext_vector_type(16))) __bf16 bf16x16;
typedef __attribute__((ext_vector_type(8)))  __bf16 bf16x8;
typedef __attribute__((ext_vector_type(2)))  __bf16 bf16x2;
typedef __attribute__((ext_vector_type(8)))  float  f32x8;
typedef int gv4i __attribute__((vector_size(16)));   // param type of async-LDS builtin

#ifndef __has_builtin
#define __has_builtin(x) 0
#endif
#if __has_builtin(__builtin_amdgcn_global_load_async_to_lds_b128) && \
    __has_builtin(__builtin_amdgcn_s_wait_asynccnt)
#define HAVE_ASYNC_LDS 1
#else
#define HAVE_ASYNC_LDS 0
#endif

static __device__ inline bf16x16 combine8(bf16x8 a, bf16x8 b) {
    return __builtin_shufflevector(a, b, 0,1,2,3,4,5,6,7,8,9,10,11,12,13,14,15);
}

static __device__ inline f32x8 wmma_bf16(bf16x16 a, bf16x16 b, f32x8 c) {
    return __builtin_amdgcn_wmma_f32_16x16x32_bf16(
        /*neg_a=*/false, a, /*neg_b=*/false, b,
        /*c_mod=*/(short)0, c, /*reuse_a=*/false, /*reuse_b=*/false);
}

// ---------------------------------------------------------------------------
// Problem constants
// ---------------------------------------------------------------------------
#define SEQ      1024
#define CACHE    1024
#define TOTAL    2048          // CACHE + SEQ
#define DIM      4096
#define HEADS    32
#define HEAD_DIM 128
#define HIDDEN   11008
#define EPS      1e-5f

// ---------------------------------------------------------------------------
// RMSNorm: fp32 in -> bf16 out (A matrix for WMMA GEMMs)
// ---------------------------------------------------------------------------
__global__ __launch_bounds__(256)
void rmsnorm_kernel(const float* __restrict__ x, const float* __restrict__ w,
                    __bf16* __restrict__ out)
{
    __shared__ float red[256];
    const int row = blockIdx.x;
    const int tid = threadIdx.x;
    const float* xr = x + (size_t)row * DIM;

    float s = 0.f;
    #pragma unroll
    for (int i = 0; i < DIM / 4 / 256; ++i) {
        float4 v = *(const float4*)(xr + (tid + i * 256) * 4);
        s += v.x * v.x + v.y * v.y + v.z * v.z + v.w * v.w;
    }
    red[tid] = s;
    __syncthreads();
    for (int st = 128; st > 0; st >>= 1) {
        if (tid < st) red[tid] += red[tid + st];
        __syncthreads();
    }
    const float inv = rsqrtf(red[0] / (float)DIM + EPS);

    __bf16* orow = out + (size_t)row * DIM;
    #pragma unroll
    for (int i = 0; i < DIM / 4 / 256; ++i) {
        const int e = tid + i * 256;
        float4 v  = *(const float4*)(xr + e * 4);
        float4 ww = *(const float4*)(w + e * 4);
        orow[e * 4 + 0] = (__bf16)(v.x * inv * ww.x);
        orow[e * 4 + 1] = (__bf16)(v.y * inv * ww.y);
        orow[e * 4 + 2] = (__bf16)(v.z * inv * ww.z);
        orow[e * 4 + 3] = (__bf16)(v.w * inv * ww.w);
    }
}

// ---------------------------------------------------------------------------
// WMMA GEMM: C[M,N](f32) = A[M,K](bf16) @ B[K,N](f32->bf16) (+ residual)
// 256 threads = 8 waves (2x4); 128x128 block tile, K step 32.
// Double-buffered LDS; A tile staged with GLOBAL_LOAD_ASYNC_TO_LDS_B128
// (ASYNCcnt) when available; B converted fp32->bf16 and stored transposed.
// ---------------------------------------------------------------------------
#define BM 128
#define BN 128
#define BK 32
#define LDS_STRIDE 40   // 32 + 8 pad (bf16 elems); keeps 16B alignment

static __device__ inline void storeB4(__bf16* sBb, int c4, int kk,
                                      float4 v0, float4 v1)
{
    // pack pairs along K so the transposed store is a 4B ds_store
    *(bf16x2*)(&sBb[(c4 + 0) * LDS_STRIDE + kk]) = bf16x2{(__bf16)v0.x, (__bf16)v1.x};
    *(bf16x2*)(&sBb[(c4 + 1) * LDS_STRIDE + kk]) = bf16x2{(__bf16)v0.y, (__bf16)v1.y};
    *(bf16x2*)(&sBb[(c4 + 2) * LDS_STRIDE + kk]) = bf16x2{(__bf16)v0.z, (__bf16)v1.z};
    *(bf16x2*)(&sBb[(c4 + 3) * LDS_STRIDE + kk]) = bf16x2{(__bf16)v0.w, (__bf16)v1.w};
}

template <bool RES>
__global__ __launch_bounds__(256)
void gemm_kernel(const __bf16* __restrict__ A, const float* __restrict__ B,
                 float* __restrict__ C, const float* __restrict__ Resid,
                 int M, int N, int K)
{
    __shared__ __bf16 sA[2][BM * LDS_STRIDE];   // [m][k]
    __shared__ __bf16 sB[2][BN * LDS_STRIDE];   // [n][k]  (transposed)

    const int tid  = threadIdx.x;
    const int lane = tid & 31;
    const int wv   = tid >> 5;
    const int wm   = wv >> 2;      // 0..1 : 64-row strip
    const int wn   = wv & 3;       // 0..3 : 32-col strip
    const int l16  = lane & 15;
    const int hf   = lane >> 4;

    const int bm = blockIdx.y * BM;
    const int bn = blockIdx.x * BN;

    // staging index assignments (constant per thread)
    const int ar0 = (tid * 2) >> 2,       ac0 = ((tid * 2) & 3) << 3;      // A chunk i=0
    const int ar1 = (tid * 2 + 1) >> 2,   ac1 = ((tid * 2 + 1) & 3) << 3;  // A chunk i=1
    const int bk0 = ((tid + 0  ) >> 5) << 1, bc0 = ((tid + 0  ) & 31) << 2;
    const int bk1 = ((tid + 256) >> 5) << 1, bc1 = ((tid + 256) & 31) << 2;

    f32x8 acc[4][2] = {};

    // ---- prologue: stage k0 = 0 into buffer 0
    {
#if HAVE_ASYNC_LDS
        __builtin_amdgcn_global_load_async_to_lds_b128(
            (gv4i*)(A + (size_t)(bm + ar0) * K + ac0),
            (gv4i*)&sA[0][ar0 * LDS_STRIDE + ac0], 0, 0);
        __builtin_amdgcn_global_load_async_to_lds_b128(
            (gv4i*)(A + (size_t)(bm + ar1) * K + ac1),
            (gv4i*)&sA[0][ar1 * LDS_STRIDE + ac1], 0, 0);
#else
        *(uint4*)&sA[0][ar0 * LDS_STRIDE + ac0] = *(const uint4*)(A + (size_t)(bm + ar0) * K + ac0);
        *(uint4*)&sA[0][ar1 * LDS_STRIDE + ac1] = *(const uint4*)(A + (size_t)(bm + ar1) * K + ac1);
#endif
        float4 v00 = *(const float4*)(B + (size_t)(bk0 + 0) * N + bn + bc0);
        float4 v01 = *(const float4*)(B + (size_t)(bk0 + 1) * N + bn + bc0);
        float4 v10 = *(const float4*)(B + (size_t)(bk1 + 0) * N + bn + bc1);
        float4 v11 = *(const float4*)(B + (size_t)(bk1 + 1) * N + bn + bc1);
        storeB4(sB[0], bc0, bk0, v00, v01);
        storeB4(sB[0], bc1, bk1, v10, v11);
#if HAVE_ASYNC_LDS
        __builtin_amdgcn_s_wait_asynccnt(0);
#endif
        __syncthreads();
    }

    int buf = 0;
    for (int k0 = 0; k0 < K; k0 += BK) {
        const int  nk   = k0 + BK;
        const bool more = nk < K;

        // ---- issue next-tile global traffic first (overlaps with WMMAs)
        float4 v00, v01, v10, v11;
#if !HAVE_ASYNC_LDS
        uint4 av0, av1;
#endif
        if (more) {
#if HAVE_ASYNC_LDS
            __builtin_amdgcn_global_load_async_to_lds_b128(
                (gv4i*)(A + (size_t)(bm + ar0) * K + nk + ac0),
                (gv4i*)&sA[buf ^ 1][ar0 * LDS_STRIDE + ac0], 0, 0);
            __builtin_amdgcn_global_load_async_to_lds_b128(
                (gv4i*)(A + (size_t)(bm + ar1) * K + nk + ac1),
                (gv4i*)&sA[buf ^ 1][ar1 * LDS_STRIDE + ac1], 0, 0);
#else
            av0 = *(const uint4*)(A + (size_t)(bm + ar0) * K + nk + ac0);
            av1 = *(const uint4*)(A + (size_t)(bm + ar1) * K + nk + ac1);
#endif
            v00 = *(const float4*)(B + (size_t)(nk + bk0 + 0) * N + bn + bc0);
            v01 = *(const float4*)(B + (size_t)(nk + bk0 + 1) * N + bn + bc0);
            v10 = *(const float4*)(B + (size_t)(nk + bk1 + 0) * N + bn + bc1);
            v11 = *(const float4*)(B + (size_t)(nk + bk1 + 1) * N + bn + bc1);
            if (nk + BK < K)
                __builtin_prefetch(B + (size_t)(nk + BK + (tid >> 3)) * N + bn + (tid & 7) * 16, 0, 0);
        }

        // ---- fragments from current buffer (DS reads precede DS stores)
        const __bf16* cA = sA[buf];
        const __bf16* cB = sB[buf];
        bf16x16 af[4], bfr[2];
        #pragma unroll
        for (int mi = 0; mi < 4; ++mi) {
            const int row = wm * 64 + mi * 16 + l16;
            bf16x8 lo = *(const bf16x8*)(&cA[row * LDS_STRIDE + hf * 8]);
            bf16x8 hi = *(const bf16x8*)(&cA[row * LDS_STRIDE + 16 + hf * 8]);
            af[mi] = combine8(lo, hi);
        }
        #pragma unroll
        for (int ni = 0; ni < 2; ++ni) {
            const int n = wn * 32 + ni * 16 + l16;
            bf16x8 lo = *(const bf16x8*)(&cB[n * LDS_STRIDE + hf * 8]);
            bf16x8 hi = *(const bf16x8*)(&cB[n * LDS_STRIDE + 16 + hf * 8]);
            bfr[ni] = combine8(lo, hi);
        }
        #pragma unroll
        for (int mi = 0; mi < 4; ++mi)
            #pragma unroll
            for (int ni = 0; ni < 2; ++ni)
                acc[mi][ni] = wmma_bf16(af[mi], bfr[ni], acc[mi][ni]);

        // ---- store staged tiles for next step
        if (more) {
#if !HAVE_ASYNC_LDS
            *(uint4*)&sA[buf ^ 1][ar0 * LDS_STRIDE + ac0] = av0;
            *(uint4*)&sA[buf ^ 1][ar1 * LDS_STRIDE + ac1] = av1;
#endif
            storeB4(sB[buf ^ 1], bc0, bk0, v00, v01);
            storeB4(sB[buf ^ 1], bc1, bk1, v10, v11);
        }
#if HAVE_ASYNC_LDS
        __builtin_amdgcn_s_wait_asynccnt(0);
#endif
        __syncthreads();
        buf ^= 1;
    }

    // ---- epilogue: D layout: VGPR r -> M = r + 8*hf, N = l16
    #pragma unroll
    for (int mi = 0; mi < 4; ++mi) {
        #pragma unroll
        for (int ni = 0; ni < 2; ++ni) {
            const int gn = bn + wn * 32 + ni * 16 + l16;
            #pragma unroll
            for (int r = 0; r < 8; ++r) {
                const int gm = bm + wm * 64 + mi * 16 + r + hf * 8;
                float v = acc[mi][ni][r];
                if (RES) v += Resid[(size_t)gm * N + gn];
                C[(size_t)gm * N + gn] = v;
            }
        }
    }
}

// ---------------------------------------------------------------------------
// Cache prep: copy fp32 caches into d_out, build bf16 K (head-major) and
// bf16 V (d-major) for attention.
// ---------------------------------------------------------------------------
__global__ __launch_bounds__(256)
void cache_prep_kernel(const float* __restrict__ kc, const float* __restrict__ vc,
                       float* __restrict__ kout, float* __restrict__ vout,
                       __bf16* __restrict__ kT, __bf16* __restrict__ vT2)
{
    const int idx = blockIdx.x * 256 + threadIdx.x;   // 1024*32*128
    const int s  = idx >> 12;
    const int hd = idx & 4095;
    const int h  = hd >> 7;
    const int d  = hd & 127;
    const float kv = kc[idx];
    const float vv = vc[idx];
    kout[idx] = kv;
    vout[idx] = vv;
    kT[((size_t)h * TOTAL + s) * HEAD_DIM + d]  = (__bf16)kv;
    vT2[((size_t)h * HEAD_DIM + d) * TOTAL + s] = (__bf16)vv;
}

// ---------------------------------------------------------------------------
// RoPE + concat finish.
// ---------------------------------------------------------------------------
__global__ __launch_bounds__(256)
void qkv_finish_kernel(const float* __restrict__ xq, const float* __restrict__ xk,
                       const float* __restrict__ xv,
                       const float* __restrict__ fcos, const float* __restrict__ fsin,
                       float* __restrict__ kout, float* __restrict__ vout,
                       __bf16* __restrict__ qT, __bf16* __restrict__ kT,
                       __bf16* __restrict__ vT2)
{
    const int i   = blockIdx.x * 256 + threadIdx.x;   // 1024*32*64
    const int s   = i >> 11;
    const int rem = i & 2047;
    const int h   = rem >> 6;
    const int dp  = rem & 63;

    const size_t base = (size_t)s * DIM + h * HEAD_DIM + dp * 2;
    const float c  = fcos[s * 64 + dp];
    const float sn = fsin[s * 64 + dp];

    const float q0 = xq[base], q1 = xq[base + 1];
    const float k0 = xk[base], k1 = xk[base + 1];
    const float v0 = xv[base], v1 = xv[base + 1];

    const float qr0 = q0 * c - q1 * sn, qr1 = q0 * sn + q1 * c;
    const float kr0 = k0 * c - k1 * sn, kr1 = k0 * sn + k1 * c;

    const size_t qi = ((size_t)h * SEQ + s) * HEAD_DIM + dp * 2;
    qT[qi]     = (__bf16)qr0;
    qT[qi + 1] = (__bf16)qr1;

    const size_t ko = (size_t)(CACHE + s) * DIM + h * HEAD_DIM + dp * 2;
    kout[ko]     = kr0;
    kout[ko + 1] = kr1;
    const size_t ki = ((size_t)h * TOTAL + CACHE + s) * HEAD_DIM + dp * 2;
    kT[ki]     = (__bf16)kr0;
    kT[ki + 1] = (__bf16)kr1;

    vout[ko]     = v0;
    vout[ko + 1] = v1;
    vT2[((size_t)h * HEAD_DIM + dp * 2 + 0) * TOTAL + CACHE + s] = (__bf16)v0;
    vT2[((size_t)h * HEAD_DIM + dp * 2 + 1) * TOTAL + CACHE + s] = (__bf16)v1;
}

// ---------------------------------------------------------------------------
// Flash attention (WMMA). One wave per (head, 16-query tile).
// All K and V fragments for a 32-key step are preloaded into registers before
// the WMMA chains so loads overlap with matrix math and softmax VALU work.
// ---------------------------------------------------------------------------
__global__ __launch_bounds__(128)
void attn_kernel(const __bf16* __restrict__ qT, const __bf16* __restrict__ kT,
                 const __bf16* __restrict__ vT2, const float* __restrict__ mask,
                 __bf16* __restrict__ ctx)
{
    __shared__ __bf16 sP[4][16 * LDS_STRIDE];   // per-wave 16x32 P tile (+pad)

    const int wv   = threadIdx.x >> 5;
    const int lane = threadIdx.x & 31;
    const int l16  = lane & 15;
    const int hf   = lane >> 4;

    const int work = blockIdx.x * 4 + wv;   // 0..2047
    const int h    = work >> 6;
    const int qb   = work & 63;
    const int q0   = qb * 16;

    const __bf16* qh = qT  + (size_t)h * SEQ * HEAD_DIM;
    const __bf16* kh = kT  + (size_t)h * TOTAL * HEAD_DIM;
    const __bf16* vh = vT2 + (size_t)h * HEAD_DIM * TOTAL;

    bf16x16 qf[4];
    #pragma unroll
    for (int ds = 0; ds < 4; ++ds) {
        const __bf16* b = qh + (size_t)(q0 + l16) * HEAD_DIM + ds * 32 + hf * 8;
        qf[ds] = combine8(*(const bf16x8*)(b), *(const bf16x8*)(b + 16));
    }

    f32x8 o[8] = {};
    float m_r[8], l_r[8];
    #pragma unroll
    for (int r = 0; r < 8; ++r) { m_r[r] = -1e30f; l_r[r] = 0.f; }
    const float scale = 0.08838834764831845f;   // 1/sqrt(128)

    for (int j0 = 0; j0 < TOTAL; j0 += 32) {
        // ---- preload all K fragments (2 key subtiles x 4 d-steps)
        bf16x16 kf[2][4];
        #pragma unroll
        for (int t = 0; t < 2; ++t) {
            const int kc = j0 + t * 16 + l16;
            #pragma unroll
            for (int ds = 0; ds < 4; ++ds) {
                const __bf16* b = kh + (size_t)kc * HEAD_DIM + ds * 32 + hf * 8;
                kf[t][ds] = combine8(*(const bf16x8*)(b), *(const bf16x8*)(b + 16));
            }
        }
        // ---- preload all V fragments (8 d-tiles); consumed after softmax
        bf16x16 vf[8];
        #pragma unroll
        for (int dt = 0; dt < 8; ++dt) {
            const __bf16* b = vh + (size_t)(dt * 16 + l16) * TOTAL + j0 + hf * 8;
            vf[dt] = combine8(*(const bf16x8*)(b), *(const bf16x8*)(b + 16));
        }

        // ---- scores S (16 x 32) as two 16x16 tiles
        f32x8 s[2] = {};
        #pragma unroll
        for (int t = 0; t < 2; ++t)
            #pragma unroll
            for (int ds = 0; ds < 4; ++ds)
                s[t] = wmma_bf16(qf[ds], kf[t][ds], s[t]);

        // ---- online softmax (rows spread over 8 VGPRs x half-wave)
        #pragma unroll
        for (int r = 0; r < 8; ++r) {
            const size_t mrow = (size_t)(q0 + r + hf * 8) * TOTAL + j0;
            float v0 = s[0][r] * scale + mask[mrow + l16];
            float v1 = s[1][r] * scale + mask[mrow + 16 + l16];
            float mx = fmaxf(v0, v1);
            #pragma unroll
            for (int off = 8; off >= 1; off >>= 1)
                mx = fmaxf(mx, __shfl_xor(mx, off, 32));
            const float mnew  = fmaxf(m_r[r], mx);
            const float alpha = __expf(m_r[r] - mnew);
            const float p0 = __expf(v0 - mnew);
            const float p1 = __expf(v1 - mnew);
            float rs = p0 + p1;
            #pragma unroll
            for (int off = 8; off >= 1; off >>= 1)
                rs += __shfl_xor(rs, off, 32);
            l_r[r] = l_r[r] * alpha + rs;
            m_r[r] = mnew;
            #pragma unroll
            for (int dt = 0; dt < 8; ++dt) o[dt][r] *= alpha;
            sP[wv][(r + hf * 8) * LDS_STRIDE + l16]      = (__bf16)p0;
            sP[wv][(r + hf * 8) * LDS_STRIDE + 16 + l16] = (__bf16)p1;
        }
        __syncthreads();

        // ---- P fragment (A layout) and O += P @ V
        bf16x8 plo = *(const bf16x8*)(&sP[wv][l16 * LDS_STRIDE + hf * 8]);
        bf16x8 phi = *(const bf16x8*)(&sP[wv][l16 * LDS_STRIDE + 16 + hf * 8]);
        bf16x16 pf = combine8(plo, phi);
        #pragma unroll
        for (int dt = 0; dt < 8; ++dt)
            o[dt] = wmma_bf16(pf, vf[dt], o[dt]);
        __syncthreads();
    }

    // ---- normalize and write context (bf16 A matrix for Wo GEMM)
    #pragma unroll
    for (int dt = 0; dt < 8; ++dt) {
        #pragma unroll
        for (int r = 0; r < 8; ++r) {
            const float v = o[dt][r] / l_r[r];
            ctx[(size_t)(q0 + r + hf * 8) * DIM + h * HEAD_DIM + dt * 16 + l16] = (__bf16)v;
        }
    }
}

// ---------------------------------------------------------------------------
// SwiGLU gate: out = silu(u) * t, bf16 output (A matrix for W2 GEMM)
// ---------------------------------------------------------------------------
__global__ __launch_bounds__(256)
void silu_mul_kernel(const float* __restrict__ u, const float* __restrict__ t,
                     __bf16* __restrict__ out, int n)
{
    const int i = blockIdx.x * 256 + threadIdx.x;
    if (i < n) {
        const float a   = u[i];
        const float sig = 1.f / (1.f + __expf(-a));
        out[i] = (__bf16)(a * sig * t[i]);
    }
}

// ---------------------------------------------------------------------------
// Host-side launcher
// ---------------------------------------------------------------------------
extern "C" void kernel_launch(void* const* d_in, const int* in_sizes, int n_in,
                              void* d_out, int out_size, void* d_ws, size_t ws_size,
                              hipStream_t stream)
{
    (void)in_sizes; (void)n_in; (void)out_size; (void)ws_size;

    const float* x       = (const float*)d_in[0];
    const float* k_cache = (const float*)d_in[1];
    const float* v_cache = (const float*)d_in[2];
    const float* fcos    = (const float*)d_in[3];
    const float* fsin    = (const float*)d_in[4];
    const float* mask    = (const float*)d_in[5];
    const float* wq      = (const float*)d_in[6];
    const float* wk      = (const float*)d_in[7];
    const float* wvv     = (const float*)d_in[8];
    const float* wo      = (const float*)d_in[9];
    const float* w1      = (const float*)d_in[10];
    const float* w2      = (const float*)d_in[11];
    const float* w3      = (const float*)d_in[12];
    const float* attn_w  = (const float*)d_in[13];
    const float* ffn_w   = (const float*)d_in[14];

    float* k_new = (float*)d_out;
    float* v_new = k_new + (size_t)TOTAL * DIM;
    float* x_out = v_new + (size_t)TOTAL * DIM;

    char* ws = (char*)d_ws;
    size_t off = 0;
    auto take = [&](size_t bytes) -> char* {
        char* p = ws + off;
        off += (bytes + 255) & ~(size_t)255;
        return p;
    };
    __bf16* h_bf   = (__bf16*)take((size_t)SEQ * DIM * 2);
    __bf16* g_bf   = (__bf16*)take((size_t)SEQ * DIM * 2);
    __bf16* ctx_bf = (__bf16*)take((size_t)SEQ * DIM * 2);
    __bf16* qT     = (__bf16*)take((size_t)HEADS * SEQ * HEAD_DIM * 2);
    __bf16* kT     = (__bf16*)take((size_t)HEADS * TOTAL * HEAD_DIM * 2);
    __bf16* vT2    = (__bf16*)take((size_t)HEADS * HEAD_DIM * TOTAL * 2);
    float*  xq     = (float*)take((size_t)SEQ * DIM * 4);
    float*  xk     = (float*)take((size_t)SEQ * DIM * 4);
    float*  xv     = (float*)take((size_t)SEQ * DIM * 4);
    float*  x1     = (float*)take((size_t)SEQ * DIM * 4);
    float*  u      = (float*)take((size_t)SEQ * HIDDEN * 4);
    float*  tg     = (float*)take((size_t)SEQ * HIDDEN * 4);
    __bf16* hb     = (__bf16*)take((size_t)SEQ * HIDDEN * 2);

    const dim3 blk256(256), blk128(128);
    const dim3 gDim(DIM / BN, SEQ / BM);      // (32, 8)
    const dim3 gHid(HIDDEN / BN, SEQ / BM);   // (86, 8)

    rmsnorm_kernel<<<SEQ, blk256, 0, stream>>>(x, attn_w, h_bf);

    gemm_kernel<false><<<gDim, blk256, 0, stream>>>(h_bf, wq,  xq, nullptr, SEQ, DIM, DIM);
    gemm_kernel<false><<<gDim, blk256, 0, stream>>>(h_bf, wk,  xk, nullptr, SEQ, DIM, DIM);
    gemm_kernel<false><<<gDim, blk256, 0, stream>>>(h_bf, wvv, xv, nullptr, SEQ, DIM, DIM);

    cache_prep_kernel<<<(CACHE * DIM) / 256, blk256, 0, stream>>>(
        k_cache, v_cache, k_new, v_new, kT, vT2);
    qkv_finish_kernel<<<(SEQ * HEADS * 64) / 256, blk256, 0, stream>>>(
        xq, xk, xv, fcos, fsin, k_new, v_new, qT, kT, vT2);

    attn_kernel<<<(HEADS * (SEQ / 16)) / 4, blk128, 0, stream>>>(qT, kT, vT2, mask, ctx_bf);

    gemm_kernel<true><<<gDim, blk256, 0, stream>>>(ctx_bf, wo, x1, x, SEQ, DIM, DIM);

    rmsnorm_kernel<<<SEQ, blk256, 0, stream>>>(x1, ffn_w, g_bf);
    gemm_kernel<false><<<gHid, blk256, 0, stream>>>(g_bf, w1, u,  nullptr, SEQ, HIDDEN, DIM);
    gemm_kernel<false><<<gHid, blk256, 0, stream>>>(g_bf, w3, tg, nullptr, SEQ, HIDDEN, DIM);
    silu_mul_kernel<<<(SEQ * HIDDEN + 255) / 256, blk256, 0, stream>>>(
        u, tg, hb, SEQ * HIDDEN);
    gemm_kernel<true><<<gDim, blk256, 0, stream>>>(hb, w2, x_out, x1, SEQ, DIM, HIDDEN);
}
